// PCT_Tokenizer_41317585388074
// MI455X (gfx1250) — matrix operand order repaired and checked by reading, not compiled, not run on hardware
//
#include <hip/hip_runtime.h>
#include <hip/hip_bf16.h>

typedef __attribute__((ext_vector_type(16))) __bf16 v16bf;
typedef __attribute__((ext_vector_type(8)))  __bf16 v8bf;
typedef __attribute__((ext_vector_type(8)))  float  v8f;
typedef unsigned short u16;

#define Bsz   1024
#define Jn    17
#define Dd    512
#define Mtok  34
#define Kcb   2048
#define DT    512
#define DDEC  32
#define DDECI 64
#define ROWS_BJ (Bsz*Jn)      // 17408
#define ROWS_BM (Bsz*Mtok)    // 34816

__device__ __forceinline__ u16 f2bf_u(float f) {
  unsigned u = __builtin_bit_cast(unsigned, f);
  u += 0x7FFFu + ((u >> 16) & 1u);
  return (u16)(u >> 16);
}

__device__ __forceinline__ float gelu_exact(float x) {
  return 0.5f * x * (1.0f + erff(x * 0.70710678118654752f));
}

__device__ __forceinline__ float block_sum(float v, float* red) {
  #pragma unroll
  for (int o = 16; o > 0; o >>= 1) v += __shfl_down(v, o);
  int lane = threadIdx.x & 31, wid = threadIdx.x >> 5;
  int nw = (blockDim.x + 31) >> 5;
  if (lane == 0) red[wid] = v;
  __syncthreads();
  float r = 0.f;
  if ((int)threadIdx.x < nw) r = red[threadIdx.x];
  if (wid == 0) {
    #pragma unroll
    for (int o = 16; o > 0; o >>= 1) r += __shfl_down(r, o);
    if (lane == 0) red[0] = r;
  }
  __syncthreads();
  r = red[0];
  __syncthreads();
  return r;
}

// ---------------- conversions (run once per call, bandwidth-trivial) --------
__global__ __launch_bounds__(256)
void cvt_kernel(const float* __restrict__ src, u16* __restrict__ dst, int n) {
  int t = blockIdx.x * blockDim.x + threadIdx.x;
  if (t < n) dst[t] = f2bf_u(src[t]);
}

// src [K,N] f32 row-major -> dst [N,K] bf16 (transposed, K-contiguous)
__global__ __launch_bounds__(256)
void cvt_tr_kernel(const float* __restrict__ src, u16* __restrict__ dst, int K_, int N_) {
  int t = blockIdx.x * blockDim.x + threadIdx.x;
  if (t >= K_ * N_) return;
  int k = t / N_, n = t - k * N_;
  dst[(size_t)n * K_ + k] = f2bf_u(src[t]);
}

// ---------------- embed ----------------
__global__ __launch_bounds__(256)
void embed_kernel(const float* __restrict__ joints, const float* __restrict__ sw,
                  const float* __restrict__ sb, const float* __restrict__ inv,
                  float* __restrict__ X) {
  int t = blockIdx.x * blockDim.x + threadIdx.x;
  if (t >= ROWS_BJ * Dd) return;
  int d = t & (Dd - 1);
  int bj = t >> 9;
  const float* jr = joints + (size_t)bj * 3;
  float val = jr[0] * sw[d] + jr[1] * sw[Dd + d] + sb[d];
  float vis = (jr[2] != 0.0f) ? 1.0f : 0.0f;
  X[t] = val * vis + inv[d] * (1.0f - vis);
}

// ---------------- LayerNorm (optional f32 and/or bf16 outputs) --------------
__global__ __launch_bounds__(256)
void ln_kernel(const float* __restrict__ X, const float* __restrict__ g,
               const float* __restrict__ b, float* __restrict__ Y,
               u16* __restrict__ Ybf, int D_) {
  __shared__ float red[8];
  size_t row = blockIdx.x;
  const float* x = X + row * D_;
  float s = 0.f;
  for (int i = threadIdx.x; i < D_; i += blockDim.x) s += x[i];
  float mean = block_sum(s, red) / (float)D_;
  float vs = 0.f;
  for (int i = threadIdx.x; i < D_; i += blockDim.x) { float d0 = x[i] - mean; vs += d0 * d0; }
  float var = block_sum(vs, red) / (float)D_;
  float rinv = rsqrtf(var + 1e-5f);
  for (int i = threadIdx.x; i < D_; i += blockDim.x) {
    float v = (x[i] - mean) * rinv * g[i] + b[i];
    if (Y)   Y[row * D_ + i] = v;
    if (Ybf) Ybf[row * D_ + i] = f2bf_u(v);
  }
}

// ---------------- token mixing over joints (f32, tiny) ----------------
__global__ __launch_bounds__(256)
void tokenmix_kernel(const float* __restrict__ Lin, const float* __restrict__ X,
                     const float* __restrict__ tw1, const float* __restrict__ tb1,
                     const float* __restrict__ tw2, const float* __restrict__ tb2,
                     float* __restrict__ XY, int D_) {
  __shared__ float s_w1[Jn * 64], s_w2[64 * Jn], s_b1[64], s_b2[Jn];
  for (int i = threadIdx.x; i < Jn * 64; i += blockDim.x) { s_w1[i] = tw1[i]; s_w2[i] = tw2[i]; }
  for (int i = threadIdx.x; i < 64; i += blockDim.x) s_b1[i] = tb1[i];
  if (threadIdx.x < Jn) s_b2[threadIdx.x] = tb2[threadIdx.x];
  __syncthreads();
  int t = blockIdx.x * blockDim.x + threadIdx.x;
  int b = t / D_, d = t - b * D_;
  size_t base = (size_t)b * Jn * D_ + d;
  float lv[Jn], o[Jn];
  #pragma unroll
  for (int j = 0; j < Jn; ++j) { lv[j] = Lin[base + (size_t)j * D_]; o[j] = s_b2[j]; }
  #pragma unroll 4
  for (int h = 0; h < 64; ++h) {
    float s = s_b1[h];
    #pragma unroll
    for (int j = 0; j < Jn; ++j) s += lv[j] * s_w1[j * 64 + h];
    s = gelu_exact(s);
    #pragma unroll
    for (int j = 0; j < Jn; ++j) o[j] += s * s_w2[h * Jn + j];
  }
  #pragma unroll
  for (int j = 0; j < Jn; ++j)
    XY[base + (size_t)j * D_] = X[base + (size_t)j * D_] + o[j];
}

// ---------------- linear over token axis, bf16 output ----------------
template <int JIN, int JOUT>
__global__ __launch_bounds__(256)
void joint_linear_kernel(const float* __restrict__ In, const float* __restrict__ W,
                         const float* __restrict__ bias, u16* __restrict__ Out, int D_) {
  __shared__ float sW[JIN * JOUT], sB[JOUT];
  for (int i = threadIdx.x; i < JIN * JOUT; i += blockDim.x) sW[i] = W[i];
  for (int i = threadIdx.x; i < JOUT; i += blockDim.x) sB[i] = bias[i];
  __syncthreads();
  int t = blockIdx.x * blockDim.x + threadIdx.x;
  int b = t / D_, d = t - b * D_;
  float lv[JIN];
  #pragma unroll
  for (int j = 0; j < JIN; ++j) lv[j] = In[(size_t)(b * JIN + j) * D_ + d];
  #pragma unroll
  for (int m = 0; m < JOUT; ++m) {
    float s = sB[m];
    #pragma unroll
    for (int j = 0; j < JIN; ++j) s += lv[j] * sW[j * JOUT + m];
    Out[(size_t)(b * JOUT + m) * D_ + d] = f2bf_u(s);
  }
}

// ---------------- WMMA bf16 GEMM ----------------
// A [M,K] bf16 row-major; Bt [N,K] bf16 (pre-transposed, K-contiguous).
// C = act(A@B + bias) (+Res).  Outputs: f32 C and/or bf16 Cbf (nullable).
// M%128==0, N%32==0, K%32==0. Workgroup tile 128x32, wave tile 16x32.
#define GBM 128
#define GBN 32
#define GBK 32
#define LSTR 40   // 32 bf16 + 16B pad; 80B row = 20-dword bank stride
__global__ __launch_bounds__(256)
void gemm_kernel(const u16* __restrict__ A, const u16* __restrict__ Bt,
                 const float* __restrict__ bias, const float* __restrict__ Res,
                 float* __restrict__ C, u16* __restrict__ Cbf,
                 int M, int N, int K, int act) {
  __shared__ __align__(16) u16 As[GBM * LSTR];
  __shared__ __align__(16) u16 Bs[GBN * LSTR];
  int tid = threadIdx.x, lane = tid & 31, wid = tid >> 5;
  int half = lane >> 4, l = lane & 15;
  int bm = blockIdx.x * GBM, bn = blockIdx.y * GBN;
  v8f acc0 = {}, acc1 = {};
  for (int k0 = 0; k0 < K; k0 += GBK) {
    #pragma unroll
    for (int i = 0; i < 2; ++i) {               // A: 128 rows x 4 chunks of 8 bf16
      int idx = tid + i * 256;
      int r = idx >> 2, c4 = idx & 3;
      *(uint4*)&As[r * LSTR + c4 * 8] =
          *(const uint4*)&A[(size_t)(bm + r) * K + k0 + c4 * 8];
    }
    if (tid < 128) {                            // B: 32 cols x 4 chunks
      int r = tid >> 2, c4 = tid & 3;
      *(uint4*)&Bs[r * LSTR + c4 * 8] =
          *(const uint4*)&Bt[(size_t)(bn + r) * K + k0 + c4 * 8];
    }
    if (k0 + GBK < K)
      __builtin_prefetch(&A[(size_t)(bm + (tid >> 1)) * K + k0 + GBK], 0, 3);
    __syncthreads();
    // A fragment: two contiguous 16B chunks per lane (ds_load_b128 x2)
    const v8bf* ap = (const v8bf*)&As[(wid * 16 + l) * LSTR];
    v8bf a0 = ap[half], a1 = ap[2 + half];
    v16bf af;
    #pragma unroll
    for (int i = 0; i < 8; ++i) { af[i] = a0[i]; af[8 + i] = a1[i]; }
    // B fragments: columns bn+l and bn+16+l; K units [h],[2+h]
    const v8bf* bp0 = (const v8bf*)&Bs[l * LSTR];
    v8bf b0 = bp0[half], b1 = bp0[2 + half];
    v16bf bf0;
    #pragma unroll
    for (int i = 0; i < 8; ++i) { bf0[i] = b0[i]; bf0[8 + i] = b1[i]; }
    const v8bf* bp1 = (const v8bf*)&Bs[(16 + l) * LSTR];
    v8bf b2 = bp1[half], b3 = bp1[2 + half];
    v16bf bf1;
    #pragma unroll
    for (int i = 0; i < 8; ++i) { bf1[i] = b2[i]; bf1[8 + i] = b3[i]; }
    acc0 = __builtin_amdgcn_wmma_f32_16x16x32_bf16(false, af, false, bf0,
                                                   (short)0, acc0, false, false);
    acc1 = __builtin_amdgcn_wmma_f32_16x16x32_bf16(false, af, false, bf1,
                                                   (short)0, acc1, false, false);
    __syncthreads();
  }
  int c0 = bn + l, c1 = bn + 16 + l;
  float bb0 = bias ? bias[c0] : 0.f;
  float bb1 = bias ? bias[c1] : 0.f;
  #pragma unroll
  for (int i = 0; i < 8; ++i) {
    int row = bm + wid * 16 + i + 8 * half;
    float v0 = acc0[i] + bb0;
    float v1 = acc1[i] + bb1;
    if (act == 1) { v0 = gelu_exact(v0); v1 = gelu_exact(v1); }
    if (Res) {
      v0 += Res[(size_t)row * N + c0];
      v1 += Res[(size_t)row * N + c1];
    }
    if (C) {
      C[(size_t)row * N + c0] = v0;
      C[(size_t)row * N + c1] = v1;
    }
    if (Cbf) {
      Cbf[(size_t)row * N + c0] = f2bf_u(v0);
      Cbf[(size_t)row * N + c1] = f2bf_u(v1);
    }
  }
}

// ---------------- codebook squared norms ----------------
__global__ __launch_bounds__(256)
void cbnorm_kernel(const float* __restrict__ cb, float* __restrict__ cbn) {
  __shared__ float red[8];
  int k = blockIdx.x;
  float s = 0.f;
  for (int d = threadIdx.x; d < DT; d += blockDim.x) {
    float v = cb[(size_t)k * DT + d];
    s += v * v;
  }
  s = block_sum(s, red);
  if (threadIdx.x == 0) cbn[k] = s;
}

// ---------------- fused distance-GEMM + argmin ----------------
// featb [N,512] bf16, cbb [2048,512] bf16 (entry-major == transposed-B layout)
#define VSTR 520  // 512 + 16B pad
__global__ __launch_bounds__(256)
void vq_kernel(const u16* __restrict__ featb, const u16* __restrict__ cbb,
               const float* __restrict__ cbn, int* __restrict__ idxo) {
  __shared__ __align__(16) u16 As[32 * VSTR];
  __shared__ __align__(16) u16 Bs[64 * LSTR];
  __shared__ float svb[32][4];
  __shared__ int   sib[32][4];
  int tid = threadIdx.x, lane = tid & 31, wid = tid >> 5;
  int wr = wid >> 2, wc = wid & 3;
  int half = lane >> 4, l = lane & 15;
  int n0 = blockIdx.x * 32;
  #pragma unroll
  for (int j = 0; j < 8; ++j) {                 // 32 rows x 64 chunks of 8 bf16
    int idx = tid + j * 256;
    int r = idx >> 6, c = idx & 63;
    *(uint4*)&As[r * VSTR + c * 8] =
        *(const uint4*)&featb[(size_t)(n0 + r) * DT + c * 8];
  }
  __syncthreads();
  float bv[8]; int bi[8];
  #pragma unroll
  for (int i = 0; i < 8; ++i) { bv[i] = 3.4e38f; bi[i] = 0; }
  const v8bf* ap = (const v8bf*)&As[(wr * 16 + l) * VSTR];
  for (int ct = 0; ct < Kcb; ct += 64) {
    v8f acc = {};
    for (int ks = 0; ks < DT; ks += 32) {
      __syncthreads();
      {                                         // 64 entries x 4 chunks
        int e = tid >> 2, c4 = tid & 3;
        *(uint4*)&Bs[e * LSTR + c4 * 8] =
            *(const uint4*)&cbb[(size_t)(ct + e) * DT + ks + c4 * 8];
      }
      if (ks + 32 < DT)
        __builtin_prefetch(&cbb[(size_t)(ct + (tid >> 2)) * DT + ks + 32], 0, 3);
      __syncthreads();
      int u = ks >> 3;
      v8bf a0 = ap[u + half], a1 = ap[u + 2 + half];
      v16bf af;
      #pragma unroll
      for (int i = 0; i < 8; ++i) { af[i] = a0[i]; af[8 + i] = a1[i]; }
      const v8bf* bp = (const v8bf*)&Bs[(wc * 16 + l) * LSTR];
      v8bf b0 = bp[half], b1 = bp[2 + half];
      v16bf bfr;
      #pragma unroll
      for (int i = 0; i < 8; ++i) { bfr[i] = b0[i]; bfr[8 + i] = b1[i]; }
      acc = __builtin_amdgcn_wmma_f32_16x16x32_bf16(false, af, false, bfr,
                                                    (short)0, acc, false, false);
    }
    int ce = ct + wc * 16 + l;
    float cn = cbn[ce];
    #pragma unroll
    for (int i = 0; i < 8; ++i) {
      float cand = cn - 2.0f * acc[i];          // + ||f||^2 is row-constant
      if (cand < bv[i]) { bv[i] = cand; bi[i] = ce; }
    }
  }
  #pragma unroll
  for (int i = 0; i < 8; ++i) {
    float v = bv[i]; int ii = bi[i];
    #pragma unroll
    for (int o = 1; o < 16; o <<= 1) {
      float ov = __shfl_xor(v, o);
      int oi = __shfl_xor(ii, o);
      if (ov < v || (ov == v && oi < ii)) { v = ov; ii = oi; }
    }
    if (l == 0) {
      int row = wr * 16 + i + 8 * half;
      svb[row][wc] = v; sib[row][wc] = ii;
    }
  }
  __syncthreads();
  if (tid < 32) {
    float v = svb[tid][0]; int ii = sib[tid][0];
    #pragma unroll
    for (int c = 1; c < 4; ++c) {
      float ov = svb[tid][c]; int oi = sib[tid][c];
      if (ov < v || (ov == v && oi < ii)) { v = ov; ii = oi; }
    }
    idxo[n0 + tid] = ii;
  }
}

// ---------------- EMA scatter ----------------
__global__ __launch_bounds__(256)
void scatter_kernel(const float* __restrict__ feat, const int* __restrict__ idx,
                    float* __restrict__ counts, float* __restrict__ dw,
                    float* __restrict__ idxf) {
  int n = blockIdx.x;
  int k = idx[n];
  if (threadIdx.x == 0) { atomicAdd(&counts[k], 1.0f); idxf[n] = (float)k; }
  for (int d = threadIdx.x; d < DT; d += blockDim.x)
    atomicAdd(&dw[(size_t)k * DT + d], feat[(size_t)n * DT + d]);
}

// ---------------- quant gather + loss ----------------
__global__ __launch_bounds__(256)
void quantloss_kernel(const float* __restrict__ feat, const float* __restrict__ cb,
                      const int* __restrict__ idx, float* __restrict__ ptf,
                      float* __restrict__ lossacc) {
  __shared__ float red[8];
  int n = blockIdx.x;
  int k = idx[n];
  float ls = 0.f;
  for (int d = threadIdx.x; d < DT; d += blockDim.x) {
    float q = cb[(size_t)k * DT + d];
    float f = feat[(size_t)n * DT + d];
    ptf[(size_t)n * DT + d] = q;
    float df = q - f;
    ls += df * df;
  }
  ls = block_sum(ls, red);
  if (threadIdx.x == 0) atomicAdd(lossacc, ls);
}

// ---------------- Laplace-smoothed cluster sizes ----------------
__global__ __launch_bounds__(256)
void cs_kernel(const float* __restrict__ ema_cs, const float* __restrict__ counts,
               float* __restrict__ cs) {
  __shared__ float red[8];
  float s = 0.f;
  for (int k = threadIdx.x; k < Kcb; k += blockDim.x) {
    float c = ema_cs[k] * 0.9f + 0.1f * counts[k];
    cs[k] = c; s += c;
  }
  float n = block_sum(s, red);
  for (int k = threadIdx.x; k < Kcb; k += blockDim.x) {
    float c = cs[k];
    cs[k] = (c + 1e-5f) / (n + (float)Kcb * 1e-5f) * n;
  }
}

__global__ __launch_bounds__(256)
void cbnew_kernel(const float* __restrict__ ema_w, const float* __restrict__ dw,
                  const float* __restrict__ cs, float* __restrict__ outcb) {
  int t = blockIdx.x * blockDim.x + threadIdx.x;
  if (t >= Kcb * DT) return;
  int k = t >> 9;
  outcb[t] = (ema_w[t] * 0.9f + 0.1f * dw[t]) / cs[k];
}

__global__ void loss_fin_kernel(const float* __restrict__ lossacc, float* __restrict__ out) {
  out[0] = lossacc[0] / ((float)ROWS_BM * (float)DT);
}

__global__ __launch_bounds__(256)
void rec_kernel(const float* __restrict__ Y, const float* __restrict__ w,
                const float* __restrict__ bvec, float* __restrict__ rec) {
  int r = blockIdx.x * blockDim.x + threadIdx.x;
  if (r >= ROWS_BJ) return;
  float s0 = bvec[0], s1 = bvec[1];
  #pragma unroll
  for (int d = 0; d < DDEC; ++d) {
    float v = Y[(size_t)r * DDEC + d];
    s0 += v * w[d * 2];
    s1 += v * w[d * 2 + 1];
  }
  rec[r * 2] = s0; rec[r * 2 + 1] = s1;
}

__global__ __launch_bounds__(256)
void zero_kernel(float* __restrict__ p, int n) {
  int t = blockIdx.x * blockDim.x + threadIdx.x;
  if (t < n) p[t] = 0.f;
}

extern "C" void kernel_launch(void* const* d_in, const int* in_sizes, int n_in,
                              void* d_out, int out_size, void* d_ws, size_t ws_size,
                              hipStream_t stream) {
  (void)in_sizes; (void)n_in; (void)out_size; (void)ws_size;
  const float* joints   = (const float*)d_in[0];
  const float* inv_tok  = (const float*)d_in[3];
  const float* start_w  = (const float*)d_in[4];
  const float* start_b  = (const float*)d_in[5];
  const float* e_ln1g = (const float*)d_in[6];
  const float* e_ln1b = (const float*)d_in[7];
  const float* e_tw1  = (const float*)d_in[8];
  const float* e_tb1  = (const float*)d_in[9];
  const float* e_tw2  = (const float*)d_in[10];
  const float* e_tb2  = (const float*)d_in[11];
  const float* e_ln2g = (const float*)d_in[12];
  const float* e_ln2b = (const float*)d_in[13];
  const float* e_cw1  = (const float*)d_in[14];
  const float* e_cb1  = (const float*)d_in[15];
  const float* e_cw2  = (const float*)d_in[16];
  const float* e_cb2  = (const float*)d_in[17];
  const float* d_ln1g = (const float*)d_in[18];
  const float* d_ln1b = (const float*)d_in[19];
  const float* d_tw1  = (const float*)d_in[20];
  const float* d_tb1  = (const float*)d_in[21];
  const float* d_tw2  = (const float*)d_in[22];
  const float* d_tb2  = (const float*)d_in[23];
  const float* d_ln2g = (const float*)d_in[24];
  const float* d_ln2b = (const float*)d_in[25];
  const float* d_cw1  = (const float*)d_in[26];
  const float* d_cb1  = (const float*)d_in[27];
  const float* d_cw2  = (const float*)d_in[28];
  const float* d_cb2  = (const float*)d_in[29];
  const float* enc_lng = (const float*)d_in[30];
  const float* enc_lnb = (const float*)d_in[31];
  const float* tok_w   = (const float*)d_in[32];
  const float* tok_b   = (const float*)d_in[33];
  const float* feat_w  = (const float*)d_in[34];
  const float* feat_b  = (const float*)d_in[35];
  const float* codebook = (const float*)d_in[36];
  const float* ema_cs   = (const float*)d_in[37];
  const float* ema_w    = (const float*)d_in[38];
  const float* dtok_w   = (const float*)d_in[39];
  const float* dtok_b   = (const float*)d_in[40];
  const float* dst_w    = (const float*)d_in[41];
  const float* dst_b    = (const float*)d_in[42];
  const float* dlng     = (const float*)d_in[43];
  const float* dlnb     = (const float*)d_in[44];
  const float* rec_w    = (const float*)d_in[45];
  const float* rec_b    = (const float*)d_in[46];

  float* out = (float*)d_out;
  float* out_rec  = out;
  float* out_idx  = out + ROWS_BJ * 2;
  float* out_loss = out + ROWS_BJ * 2 + ROWS_BM;
  float* out_ptf  = out_loss + 1;
  float* out_cbn  = out_ptf + (size_t)ROWS_BM * DT;

  // ---- workspace carving (float units; every region 16B-aligned) ----
  const size_t SZ_XJD = (size_t)ROWS_BJ * Dd;   // 8,912,896
  const size_t SZ_TMD = (size_t)ROWS_BM * DT;   // 17,825,792
  float* base = (float*)d_ws;
  size_t cur = 0;
  auto carve_f = [&](size_t n) { float* p = base + cur; cur += (n + 7) & ~(size_t)7; return p; };
  float* w_x    = carve_f(SZ_XJD);
  float* w_xy   = carve_f(SZ_XJD);
  float* w_ln   = carve_f(SZ_XJD);
  float* w_feat = carve_f(SZ_TMD);
  float* w_cbn  = carve_f(Kcb);
  int*   w_idx  = (int*)carve_f(ROWS_BM);
  float* w_cnt  = carve_f(Kcb);
  float* w_dw   = carve_f((size_t)Kcb * DT);
  float* w_cs   = carve_f(Kcb);
  float* w_loss = carve_f(8);
  auto carve_h = [&](size_t n) { u16* p = (u16*)(base + cur); cur += (n + 15) / 2 & ~(size_t)7; cur += 8; return p; };
  u16* b_ln   = carve_h(SZ_XJD);                 // bf16 LN / activations
  u16* b_h1   = carve_h(SZ_XJD);
  u16* b_t    = carve_h(SZ_TMD);
  u16* b_feat = carve_h(SZ_TMD);
  u16* b_y0   = carve_h(SZ_XJD);
  u16* b_cb   = carve_h((size_t)Kcb * DT);
  u16* bt_cw1 = carve_h((size_t)4 * Dd * Dd);    // pre-transposed weights
  u16* bt_cw2 = carve_h((size_t)4 * Dd * Dd);
  u16* bt_fw  = carve_h((size_t)Dd * DT);
  u16* bt_dsw = carve_h((size_t)DT * DDEC);
  u16* bt_dc1 = carve_h((size_t)DDEC * DDECI);
  u16* bt_dc2 = carve_h((size_t)DDECI * DDEC);

  // ---- one-time (per call) weight conversions ----
  for (int i = 0; i < 4; ++i) {
    cvt_tr_kernel<<<(Dd * Dd + 255) / 256, 256, 0, stream>>>(
        e_cw1 + (size_t)i * Dd * Dd, bt_cw1 + (size_t)i * Dd * Dd, Dd, Dd);
    cvt_tr_kernel<<<(Dd * Dd + 255) / 256, 256, 0, stream>>>(
        e_cw2 + (size_t)i * Dd * Dd, bt_cw2 + (size_t)i * Dd * Dd, Dd, Dd);
  }
  cvt_tr_kernel<<<(Dd * DT + 255) / 256, 256, 0, stream>>>(feat_w, bt_fw, Dd, DT);
  cvt_tr_kernel<<<(DT * DDEC + 255) / 256, 256, 0, stream>>>(dst_w, bt_dsw, DT, DDEC);
  cvt_tr_kernel<<<(DDEC * DDECI + 255) / 256, 256, 0, stream>>>(d_cw1, bt_dc1, DDEC, DDECI);
  cvt_tr_kernel<<<(DDECI * DDEC + 255) / 256, 256, 0, stream>>>(d_cw2, bt_dc2, DDECI, DDEC);
  cvt_kernel<<<(Kcb * DT + 255) / 256, 256, 0, stream>>>(codebook, b_cb, Kcb * DT);

  // ---- encoder ----
  embed_kernel<<<(ROWS_BJ * Dd + 255) / 256, 256, 0, stream>>>(joints, start_w, start_b, inv_tok, w_x);
  for (int i = 0; i < 4; ++i) {
    ln_kernel<<<ROWS_BJ, 256, 0, stream>>>(w_x, e_ln1g + i * Dd, e_ln1b + i * Dd, w_ln, nullptr, Dd);
    tokenmix_kernel<<<(Bsz * Dd) / 256, 256, 0, stream>>>(w_ln, w_x,
        e_tw1 + i * Jn * 64, e_tb1 + i * 64, e_tw2 + i * 64 * Jn, e_tb2 + i * Jn, w_xy, Dd);
    ln_kernel<<<ROWS_BJ, 256, 0, stream>>>(w_xy, e_ln2g + i * Dd, e_ln2b + i * Dd, nullptr, b_ln, Dd);
    gemm_kernel<<<dim3(ROWS_BJ / GBM, Dd / GBN), 256, 0, stream>>>(
        b_ln, bt_cw1 + (size_t)i * Dd * Dd, e_cb1 + i * Dd, nullptr,
        nullptr, b_h1, ROWS_BJ, Dd, Dd, 1);
    gemm_kernel<<<dim3(ROWS_BJ / GBM, Dd / GBN), 256, 0, stream>>>(
        b_h1, bt_cw2 + (size_t)i * Dd * Dd, e_cb2 + i * Dd, w_xy,
        w_x, nullptr, ROWS_BJ, Dd, Dd, 0);
  }
  ln_kernel<<<ROWS_BJ, 256, 0, stream>>>(w_x, enc_lng, enc_lnb, w_ln, nullptr, Dd);
  joint_linear_kernel<Jn, Mtok><<<(Bsz * Dd) / 256, 256, 0, stream>>>(w_ln, tok_w, tok_b, b_t, Dd);
  gemm_kernel<<<dim3(ROWS_BM / GBM, DT / GBN), 256, 0, stream>>>(
      b_t, bt_fw, feat_b, nullptr, w_feat, b_feat, ROWS_BM, DT, Dd, 0);

  // ---- VQ + EMA ----
  cbnorm_kernel<<<Kcb, 256, 0, stream>>>(codebook, w_cbn);
  zero_kernel<<<(Kcb + 255) / 256, 256, 0, stream>>>(w_cnt, Kcb);
  zero_kernel<<<(Kcb * DT + 255) / 256, 256, 0, stream>>>(w_dw, Kcb * DT);
  zero_kernel<<<1, 256, 0, stream>>>(w_loss, 1);
  vq_kernel<<<ROWS_BM / 32, 256, 0, stream>>>(b_feat, b_cb, w_cbn, w_idx);
  scatter_kernel<<<ROWS_BM, 256, 0, stream>>>(w_feat, w_idx, w_cnt, w_dw, out_idx);
  quantloss_kernel<<<ROWS_BM, 256, 0, stream>>>(w_feat, codebook, w_idx, out_ptf, w_loss);
  cs_kernel<<<1, 256, 0, stream>>>(ema_cs, w_cnt, w_cs);
  cbnew_kernel<<<(Kcb * DT + 255) / 256, 256, 0, stream>>>(ema_w, w_dw, w_cs, out_cbn);
  loss_fin_kernel<<<1, 1, 0, stream>>>(w_loss, out_loss);

  // ---- decoder ----
  joint_linear_kernel<Mtok, Jn><<<(Bsz * DT) / 256, 256, 0, stream>>>(out_ptf, dtok_w, dtok_b, b_y0, DT);
  gemm_kernel<<<dim3(ROWS_BJ / GBM, DDEC / GBN), 256, 0, stream>>>(
      b_y0, bt_dsw, dst_b, nullptr, w_xy, nullptr, ROWS_BJ, DDEC, DT, 0);
  ln_kernel<<<ROWS_BJ, 256, 0, stream>>>(w_xy, d_ln1g, d_ln1b, w_ln, nullptr, DDEC);
  tokenmix_kernel<<<(Bsz * DDEC) / 256, 256, 0, stream>>>(w_ln, w_xy,
      d_tw1, d_tb1, d_tw2, d_tb2, w_x, DDEC);
  ln_kernel<<<ROWS_BJ, 256, 0, stream>>>(w_x, d_ln2g, d_ln2b, nullptr, b_ln, DDEC);
  gemm_kernel<<<dim3(ROWS_BJ / GBM, DDECI / GBN), 256, 0, stream>>>(
      b_ln, bt_dc1, d_cb1, nullptr, nullptr, b_h1, ROWS_BJ, DDECI, DDEC, 1);
  gemm_kernel<<<dim3(ROWS_BJ / GBM, DDEC / GBN), 256, 0, stream>>>(
      b_h1, bt_dc2, d_cb2, w_x, w_xy, nullptr, ROWS_BJ, DDEC, DDECI, 0);
  ln_kernel<<<ROWS_BJ, 256, 0, stream>>>(w_xy, dlng, dlnb, w_ln, nullptr, DDEC);
  rec_kernel<<<(ROWS_BJ + 255) / 256, 256, 0, stream>>>(w_ln, rec_w, rec_b, out_rec);
}
